// model_LSTM_classifier_58402965291684
// MI455X (gfx1250) — compile-verified, matrix operand order checked by reading, not run on hardware
//
#include <hip/hip_runtime.h>
#include <hip/hip_bf16.h>

typedef __bf16 bf16_t;
typedef __attribute__((ext_vector_type(16))) __bf16 v16bf;
typedef __attribute__((ext_vector_type(8)))  __bf16 v8bf;
typedef __attribute__((ext_vector_type(8)))  float  v8f;
typedef __attribute__((ext_vector_type(4)))  float  v4f;

#define B_SZ   32
#define T_SZ   2048
#define HDIM   1024
#define NBLOCKS 32
#define TPB    128            // 4 waves/block; 128 waves = 2 Mtiles x 64 Ntiles
#define BH     (B_SZ*HDIM)

// workspace layout (bytes)
#define WS_HBUF   0                              // float hbuf[3][32][1024]   (384 KB)
#define WS_HBF16  (3*BH*4)                       // bf16  hbf [3][32][1024]   (192 KB)
#define WS_SYNC   (WS_HBF16 + 3*BH*2)            // unsigned sync[2] {counter, generation}
#define WS_PACKED (WS_SYNC + 256)                // bf16 packedW[2][64][4][32][32][16] (16 MB)
#define PACKED_ELEMS (2u*4u*HDIM*HDIM)           // 8388608 bf16

// ---------------------------------------------------------------------------
// Pack W_ih / W_hh (f32 [4H,1024]) into bf16 WMMA-B fragment order:
//   packed[((mat*64+nt)*4+g)*32*512 + kb*512 + lane*16 + j]
//     = W[mat][g*1024 + nt*16 + (lane&15)][kb*32 + 8*(lane>>4) + (j&7) + ((j>>3)<<4)]
// so each wave's per-k-step B load is one contiguous 32B per lane.
// ---------------------------------------------------------------------------
__global__ void pack_w_kernel(const float* __restrict__ Wih,
                              const float* __restrict__ Whh,
                              bf16_t* __restrict__ packed) {
  for (unsigned d = blockIdx.x * blockDim.x + threadIdx.x; d < PACKED_ELEMS;
       d += gridDim.x * blockDim.x) {
    int j   =  d        & 15;
    int l   = (d >> 4)  & 31;
    int kb  = (d >> 9)  & 31;
    int g   = (d >> 14) & 3;
    int nt  = (d >> 16) & 63;
    int mat = (int)(d >> 22);
    int row = g * HDIM + nt * 16 + (l & 15);
    int k   = kb * 32 + ((l >> 4) << 3) + (j & 7) + ((j >> 3) << 4);
    const float* W = mat ? Whh : Wih;
    packed[d] = (bf16_t)W[(size_t)row * HDIM + k];
  }
}

// zero h buffers (f32 + bf16) and the sync words
__global__ void init_ws_kernel(unsigned* __restrict__ ws_u32) {
  const unsigned nwords = (WS_SYNC + 8) / 4;  // everything up to & incl. sync[2]
  for (unsigned i = blockIdx.x * blockDim.x + threadIdx.x; i < nwords;
       i += gridDim.x * blockDim.x)
    ws_u32[i] = 0u;
}

// ---------------------------------------------------------------------------
__device__ __forceinline__ float sigmoid_f(float x) { return 1.0f / (1.0f + __expf(-x)); }
__device__ __forceinline__ float tanh_f(float x)    { return 2.0f / (1.0f + __expf(-2.0f * x)) - 1.0f; }

// bf16 A fragment from f32 row (emb): 4x b128 loads + pack-cvt
__device__ __forceinline__ v16bf load_a_f32(const float* __restrict__ rowp, int k0, int half8) {
  const float* p0 = rowp + k0 + half8;
  const float* p1 = p0 + 16;
  v4f f0 = *(const v4f*)(p0);
  v4f f1 = *(const v4f*)(p0 + 4);
  v4f f2 = *(const v4f*)(p1);
  v4f f3 = *(const v4f*)(p1 + 4);
  v16bf a;
#pragma unroll
  for (int c = 0; c < 4; ++c) {
    a[c]      = (bf16_t)f0[c];
    a[4 + c]  = (bf16_t)f1[c];
    a[8 + c]  = (bf16_t)f2[c];
    a[12 + c] = (bf16_t)f3[c];
  }
  return a;
}

// bf16 A fragment from bf16 row (h state): two b128 loads, no conversion
__device__ __forceinline__ v16bf load_a_bf16(const bf16_t* __restrict__ rowp, int k0, int half8) {
  v8bf x0 = *(const v8bf*)(rowp + k0 + half8);
  v8bf x1 = *(const v8bf*)(rowp + k0 + half8 + 16);
  v16bf a;
#pragma unroll
  for (int c = 0; c < 8; ++c) { a[c] = x0[c]; a[8 + c] = x1[c]; }
  return a;
}

// ---------------------------------------------------------------------------
// Persistent LSTM scan. Each wave owns a 16x16 (batch x hidden) tile of h/c;
// c lives in registers for all 2048 steps. Split-barrier software pipeline:
// inside the arrive->wait window of step t's grid barrier we run
//   (a) the x-part GEMM of step t+1,
//   (b) L2 prefetch of the t+2 embedding rows (HBM-latency hiding),
//   (c) the logits of step t-1 (h is triple-buffered so this is race-free).
// ---------------------------------------------------------------------------
__global__ __launch_bounds__(TPB, 1) void lstm_persistent_kernel(
    const int* __restrict__ seq, const float* __restrict__ emb,
    const bf16_t* __restrict__ packedW, const float* __restrict__ b_ih,
    const float* __restrict__ b_hh, const float* __restrict__ W_score,
    float* __restrict__ hbuf, bf16_t* __restrict__ hbf,
    unsigned* __restrict__ sync_ws, float* __restrict__ out) {
  const int tid   = threadIdx.x;
  const int lane  = tid & 31;
  const int wv    = blockIdx.x * (TPB / 32) + (tid >> 5);  // 0..127
  const int mt    = wv & 1;                                // batch tile (0/1)
  const int nt    = wv >> 1;                               // hidden tile (0..63)
  const int l16   = lane & 15;
  const int half8 = (lane >> 4) << 3;

  const int arow = mt * 16 + l16;   // batch row this lane feeds into A
  const int ncol = nt * 16 + l16;   // hidden column of this lane's accumulators

  unsigned* cnt = sync_ws;
  unsigned* gen = sync_ws + 1;

  float bias[4];
#pragma unroll
  for (int g = 0; g < 4; ++g) bias[g] = b_ih[g * HDIM + ncol] + b_hh[g * HDIM + ncol];

  const bf16_t* Bi[4];
  const bf16_t* Bh[4];
#pragma unroll
  for (int g = 0; g < 4; ++g) {
    Bi[g] = packedW + (size_t)((0 * 64 + nt) * 4 + g) * (32 * 512) + lane * 16;
    Bh[g] = packedW + (size_t)((1 * 64 + nt) * 4 + g) * (32 * 512) + lane * 16;
  }

  // logits constants for this thread (block handles batch row == blockIdx.x)
  const int ob = blockIdx.x;
  float ws0[8], ws1[8];
#pragma unroll
  for (int c = 0; c < 8; ++c) {
    ws0[c] = W_score[tid * 8 + c];
    ws1[c] = W_score[HDIM + tid * 8 + c];
  }
  __shared__ float red[8];

  float creg[8];
#pragma unroll
  for (int r = 0; r < 8; ++r) creg[r] = 0.0f;

  const int* seqrow = seq + arow * T_SZ;

  // ---- prologue: prefetch t=0/1 rows, then x-part GEMM for t = 0 ----
  {
    const float* pf0 = emb + (size_t)seqrow[0] * HDIM + (size_t)(lane >> 4) * 512;
    const float* pf1 = emb + (size_t)seqrow[1] * HDIM + (size_t)(lane >> 4) * 512;
#pragma unroll
    for (int p = 0; p < 16; ++p) {
      __builtin_prefetch(pf0 + p * 32, 0, 3);
      __builtin_prefetch(pf1 + p * 32, 0, 3);
    }
  }
  v8f accx[4];
#pragma unroll
  for (int g = 0; g < 4; ++g)
#pragma unroll
    for (int r = 0; r < 8; ++r) accx[g][r] = bias[g];
  {
    const float* xrow = emb + (size_t)seqrow[0] * HDIM;
    for (int kb = 0; kb < 32; ++kb) {
      v16bf a = load_a_f32(xrow, kb * 32, half8);
#pragma unroll
      for (int g = 0; g < 4; ++g) {
        v16bf bf = *(const v16bf*)(Bi[g] + kb * 512);
        accx[g] = __builtin_amdgcn_wmma_f32_16x16x32_bf16(false, a, false, bf, (short)0,
                                                          accx[g], false, false);
      }
    }
  }

  for (int t = 0; t < T_SZ; ++t) {
    const int wr = t % 3;            // h_t written here
    const int rd = (t + 2) % 3;      // h_{t-1} read from here
    v8f acc[4];
#pragma unroll
    for (int g = 0; g < 4; ++g) acc[g] = accx[g];

    // ---- h-part: gates += h_{t-1} @ W_hh^T  (bf16 copy, no conversion) ----
    const bf16_t* hrow = hbf + (size_t)rd * BH + (size_t)arow * HDIM;
    for (int kb = 0; kb < 32; ++kb) {
      v16bf a = load_a_bf16(hrow, kb * 32, half8);
#pragma unroll
      for (int g = 0; g < 4; ++g) {
        v16bf bf = *(const v16bf*)(Bh[g] + kb * 512);
        acc[g] = __builtin_amdgcn_wmma_f32_16x16x32_bf16(false, a, false, bf, (short)0,
                                                         acc[g], false, false);
      }
    }

    // ---- pointwise cell update; write h_t (f32 for logits + bf16 for GEMM) ----
    float*  hnf = hbuf + (size_t)wr * BH;
    bf16_t* hnb = hbf  + (size_t)wr * BH;
#pragma unroll
    for (int r = 0; r < 8; ++r) {
      float ig = sigmoid_f(acc[0][r]);
      float fg = sigmoid_f(acc[1][r]);
      float gg = tanh_f(acc[2][r]);
      float og = sigmoid_f(acc[3][r]);
      float c  = fg * creg[r] + ig * gg;
      creg[r]  = c;
      float h  = og * tanh_f(c);
      int   b  = mt * 16 + r + half8;  // C/D layout: VGPR r -> M=r (lanes0-15), M=r+8
      hnf[(size_t)b * HDIM + ncol] = h;
      hnb[(size_t)b * HDIM + ncol] = (bf16_t)h;
    }

    // ---- barrier arrive ----
    __threadfence();
    __syncthreads();
    unsigned my_gen = 0u;
    if (tid == 0) {
      my_gen = __hip_atomic_load(gen, __ATOMIC_RELAXED, __HIP_MEMORY_SCOPE_AGENT);
      unsigned arrived =
          __hip_atomic_fetch_add(cnt, 1u, __ATOMIC_ACQ_REL, __HIP_MEMORY_SCOPE_AGENT);
      if (arrived == NBLOCKS - 1) {
        __hip_atomic_store(cnt, 0u, __ATOMIC_RELAXED, __HIP_MEMORY_SCOPE_AGENT);
        __hip_atomic_fetch_add(gen, 1u, __ATOMIC_RELEASE, __HIP_MEMORY_SCOPE_AGENT);
      }
    }

    // ==== arrive->wait window: independent work off the critical path ====

    // prefetch t+2 embedding rows into L2 (global_prefetch; full HBM-latency hide)
    if (t + 2 < T_SZ) {
      const float* pf = emb + (size_t)seqrow[t + 2] * HDIM + (size_t)(lane >> 4) * 512;
#pragma unroll
      for (int p = 0; p < 16; ++p) __builtin_prefetch(pf + p * 32, 0, 3);
    }

    // x-part GEMM for step t+1 (depends only on emb/seq)
    if (t + 1 < T_SZ) {
#pragma unroll
      for (int g = 0; g < 4; ++g)
#pragma unroll
        for (int r = 0; r < 8; ++r) accx[g][r] = bias[g];
      const float* xrow = emb + (size_t)seqrow[t + 1] * HDIM;
      for (int kb = 0; kb < 32; ++kb) {
        v16bf a = load_a_f32(xrow, kb * 32, half8);
#pragma unroll
        for (int g = 0; g < 4; ++g) {
          v16bf bf = *(const v16bf*)(Bi[g] + kb * 512);
          accx[g] = __builtin_amdgcn_wmma_f32_16x16x32_bf16(false, a, false, bf, (short)0,
                                                            accx[g], false, false);
        }
      }
    }

    // logits for step t-1 (buffer (t-1)%3 is stable until step t+2)
    if (t > 0) {
      const float* hb = hbuf + (size_t)((t + 2) % 3) * BH + (size_t)ob * HDIM;
      float p0 = 0.0f, p1 = 0.0f;
#pragma unroll
      for (int c2 = 0; c2 < 8; ++c2) {
        float hv = hb[tid * 8 + c2];
        hv = hv > 0.0f ? hv : 0.0f;
        p0 += hv * ws0[c2];
        p1 += hv * ws1[c2];
      }
#pragma unroll
      for (int s = 16; s > 0; s >>= 1) {
        p0 += __shfl_xor(p0, s, 32);
        p1 += __shfl_xor(p1, s, 32);
      }
      if (lane == 0) { red[tid >> 5] = p0; red[4 + (tid >> 5)] = p1; }
      __syncthreads();
      if (tid == 0) {
        out[((size_t)ob * T_SZ + (t - 1)) * 2 + 0] = red[0] + red[1] + red[2] + red[3];
        out[((size_t)ob * T_SZ + (t - 1)) * 2 + 1] = red[4] + red[5] + red[6] + red[7];
      }
    }

    // ---- barrier wait ----
    if (tid == 0) {
      while (__hip_atomic_load(gen, __ATOMIC_ACQUIRE, __HIP_MEMORY_SCOPE_AGENT) == my_gen)
        __builtin_amdgcn_s_sleep(1);
    }
    __syncthreads();
    __threadfence();
  }

  // ---- epilogue: logits for the final step ----
  {
    const float* hb = hbuf + (size_t)((T_SZ - 1) % 3) * BH + (size_t)ob * HDIM;
    float p0 = 0.0f, p1 = 0.0f;
#pragma unroll
    for (int c2 = 0; c2 < 8; ++c2) {
      float hv = hb[tid * 8 + c2];
      hv = hv > 0.0f ? hv : 0.0f;
      p0 += hv * ws0[c2];
      p1 += hv * ws1[c2];
    }
#pragma unroll
    for (int s = 16; s > 0; s >>= 1) {
      p0 += __shfl_xor(p0, s, 32);
      p1 += __shfl_xor(p1, s, 32);
    }
    if (lane == 0) { red[tid >> 5] = p0; red[4 + (tid >> 5)] = p1; }
    __syncthreads();
    if (tid == 0) {
      out[((size_t)ob * T_SZ + (T_SZ - 1)) * 2 + 0] = red[0] + red[1] + red[2] + red[3];
      out[((size_t)ob * T_SZ + (T_SZ - 1)) * 2 + 1] = red[4] + red[5] + red[6] + red[7];
    }
  }
}

// ---------------------------------------------------------------------------
extern "C" void kernel_launch(void* const* d_in, const int* in_sizes, int n_in,
                              void* d_out, int out_size, void* d_ws, size_t ws_size,
                              hipStream_t stream) {
  (void)in_sizes; (void)n_in; (void)out_size; (void)ws_size;
  const int*   seq     = (const int*)d_in[0];
  const float* emb     = (const float*)d_in[1];
  const float* W_ih    = (const float*)d_in[2];
  const float* W_hh    = (const float*)d_in[3];
  const float* b_ih    = (const float*)d_in[4];
  const float* b_hh    = (const float*)d_in[5];
  const float* W_score = (const float*)d_in[6];
  float*       out     = (float*)d_out;

  char*     ws     = (char*)d_ws;
  float*    hbuf   = (float*)(ws + WS_HBUF);
  bf16_t*   hbf    = (bf16_t*)(ws + WS_HBF16);
  unsigned* sync_p = (unsigned*)(ws + WS_SYNC);
  bf16_t*   packed = (bf16_t*)(ws + WS_PACKED);

  init_ws_kernel<<<600, 256, 0, stream>>>((unsigned*)d_ws);
  pack_w_kernel<<<1024, 256, 0, stream>>>(W_ih, W_hh, packed);
  lstm_persistent_kernel<<<NBLOCKS, TPB, 0, stream>>>(seq, emb, packed, b_ih, b_hh,
                                                      W_score, hbuf, hbf, sync_p, out);
}
